// SelfAttention_14748917694967
// MI455X (gfx1250) — matrix operand order repaired and checked by reading, not compiled, not run on hardware
//
#include <hip/hip_runtime.h>
#include <hip/hip_bf16.h>

typedef __attribute__((ext_vector_type(16))) __bf16 v16bf;
typedef __attribute__((ext_vector_type(8)))  __bf16 v8bf;
typedef __attribute__((ext_vector_type(8)))  float  v8f;

#define B_ 4
#define S_ 2048
#define E_ 1024
#define D_ 128

static __device__ __forceinline__ v16bf cat8(v8bf lo, v8bf hi)
{
    return __builtin_shufflevector(lo, hi, 0, 1, 2, 3, 4, 5, 6, 7,
                                           8, 9, 10, 11, 12, 13, 14, 15);
}

// ---------------------------------------------------------------------------
// Prepass: W (E x D, fp32 row-major) -> W^T (D x E, bf16).  Makes every
// stage-1 B-operand a single aligned 32-byte contiguous load.
// ---------------------------------------------------------------------------
__global__ __launch_bounds__(256)
void convert_w_kernel(const float* __restrict__ W, __bf16* __restrict__ Wt)
{
    int idx = blockIdx.x * blockDim.x + threadIdx.x;     // over D*E outputs
    if (idx >= E_ * D_) return;
    int d = idx >> 10;                                    // idx / E_
    int e = idx & (E_ - 1);
    Wt[idx] = (__bf16)W[(size_t)e * D_ + d];
}

// ---------------------------------------------------------------------------
// Stage 1: QKV projection.  grid = (B*S)/16 blocks, 256 threads (8 waves).
// The 16x32 x tile is cooperatively converted to bf16 in double-buffered LDS
// (one barrier per k-step); each wave builds its A operand with two
// ds_load_b128 and feeds 3 WMMAs per step (shared A across Q/K/V).
// Q gets the fused 1/128 score scale (reference divides by sqrt(d) twice).
// Q,K stored [b*s][d] bf16; V stored transposed [b][d][s] bf16.
// ---------------------------------------------------------------------------
__global__ __launch_bounds__(256)
void qkv_proj_kernel(const float* __restrict__ x,
                     const __bf16* __restrict__ Wtq, const float* __restrict__ bq,
                     const __bf16* __restrict__ Wtk, const float* __restrict__ bk,
                     const __bf16* __restrict__ Wtv, const float* __restrict__ bv,
                     __bf16* __restrict__ Qw, __bf16* __restrict__ Kw,
                     __bf16* __restrict__ Vt)
{
    __shared__ __bf16 xt[2][16 * 32];                    // double-buffered tile

    const int lane = threadIdx.x & 31;
    const int wave = threadIdx.x >> 5;                   // 0..7 -> N tile of D
    const int row0 = blockIdx.x * 16;                    // 16-row tile of B*S
    const int m    = lane & 15;
    const int hi   = lane >> 4;
    const int kb8  = hi * 8;                             // unused; A operand also uses kb16
    const int kb16 = hi * 16;                            // B-operand K base
    const int ncol = wave * 16 + m;                      // output column

    const int f_row0 = threadIdx.x >> 5;
    const int f_col0 = threadIdx.x & 31;
    const int f_row1 = (threadIdx.x + 256) >> 5;

    const __bf16* wqp = Wtq + (size_t)ncol * E_ + kb16;
    const __bf16* wkp = Wtk + (size_t)ncol * E_ + kb16;
    const __bf16* wvp = Wtv + (size_t)ncol * E_ + kb16;

    v8f cq = {}, ck = {}, cv = {};

    for (int i = 0; i < E_ / 32; ++i) {
        const int kk  = i * 32;
        const int buf = i & 1;
        // ---- cooperative load+convert of x tile (coalesced fp32 reads) ----
        xt[buf][threadIdx.x] =
            (__bf16)x[(size_t)(row0 + f_row0) * E_ + kk + f_col0];
        xt[buf][threadIdx.x + 256] =
            (__bf16)x[(size_t)(row0 + f_row1) * E_ + kk + f_col0];
        __syncthreads();

        // ---- A operand from LDS: two aligned 16B runs ----
        v8bf alo = *(const v8bf*)&xt[buf][m * 32 + kb16];
        v8bf ahi = *(const v8bf*)&xt[buf][m * 32 + kb16 + 8];
        v16bf a  = cat8(alo, ahi);

        // ---- B operands: one aligned 32B contiguous load each ----
        v16bf bqv = *(const v16bf*)(wqp + kk);
        v16bf bkv = *(const v16bf*)(wkp + kk);
        v16bf bvv = *(const v16bf*)(wvp + kk);

        cq = __builtin_amdgcn_wmma_f32_16x16x32_bf16(false, a, false, bqv, (short)0, cq, false, false);
        ck = __builtin_amdgcn_wmma_f32_16x16x32_bf16(false, a, false, bkv, (short)0, ck, false, false);
        cv = __builtin_amdgcn_wmma_f32_16x16x32_bf16(false, a, false, bvv, (short)0, cv, false, false);
        // single barrier per step is safe with double buffering: this wave's
        // reads of buf are consumed (WMMA waits) before it writes buf^1.
    }

    const float qb = bq[ncol], kbv = bk[ncol], vb = bv[ncol];
    // C/D layout: reg r -> row (r + 8*hi), col = lane&15 (= ncol here)
#pragma unroll
    for (int r = 0; r < 8; ++r) {
        int srow = row0 + r + 8 * hi;                    // row in [0, B*S)
        float qv = (cq[r] + qb) * (1.0f / 128.0f);       // fused 1/sqrt(d)^2
        float kv = ck[r] + kbv;
        float vv = cv[r] + vb;
        Qw[(size_t)srow * D_ + ncol] = (__bf16)qv;
        Kw[(size_t)srow * D_ + ncol] = (__bf16)kv;
        int b = srow / S_, s = srow - b * S_;
        Vt[((size_t)b * D_ + ncol) * S_ + s] = (__bf16)vv;   // transposed
    }
}

// ---------------------------------------------------------------------------
// Stage 2: KV-split flash attention.  One block (4 waves) per 16-row Q tile;
// wave w processes KV tiles j = w*32, w*32+128, ... (16 iterations each),
// maintaining private online-softmax state.  Partial (m, l, O) merged across
// waves through LDS with the standard rescale.  grid = 512, block = 128.
// ---------------------------------------------------------------------------
#define KV_WAVES 4
__global__ __launch_bounds__(32 * KV_WAVES)
void flash_attn_kernel(const __bf16* __restrict__ Qw,
                       const __bf16* __restrict__ Kw,
                       const __bf16* __restrict__ Vt,
                       float* __restrict__ out)
{
    __shared__ float  ols[KV_WAVES][16][D_];             // 32 KB partial O
    __shared__ float  mls[KV_WAVES][2][16];              // partial m / l
    __shared__ __bf16 plds[KV_WAVES][16 * 32];           // 4 KB P staging

    const int lane = threadIdx.x & 31;
    const int wave = threadIdx.x >> 5;
    const int tile = blockIdx.x;                         // 0 .. B*S/16-1
    const int b    = tile / (S_ / 16);
    const int qt   = tile - b * (S_ / 16);
    const int m    = lane & 15;
    const int hi   = lane >> 4;
    const int kb8  = hi * 8;
    const int kb16 = hi * 16;

    // Q A-operands: 4 chunks of 32 over D=128, loaded once (b128 pairs).
    v16bf qa[4];
    const __bf16* qrow = Qw + ((size_t)b * S_ + qt * 16 + m) * D_;
#pragma unroll
    for (int c = 0; c < 4; ++c) {
        v8bf lo = *(const v8bf*)(qrow + c * 32 + kb16);
        v8bf hv = *(const v8bf*)(qrow + c * 32 + kb16 + 8);
        qa[c] = cat8(lo, hv);
    }

    const __bf16* krow0 = Kw + ((size_t)b * S_ + m) * D_ + kb16;       // + j*D_
    const __bf16* krow1 = Kw + ((size_t)b * S_ + 16 + m) * D_ + kb16;  // + j*D_

    v8f o[8];
#pragma unroll
    for (int n = 0; n < 8; ++n) o[n] = (v8f){};
    float mrow[8], lrow[8];
#pragma unroll
    for (int r = 0; r < 8; ++r) { mrow[r] = -1e30f; lrow[r] = 0.0f; }

    // interleaved KV stripes: identical trip count (16) for every wave
    for (int j = wave * 32; j < S_; j += 32 * KV_WAVES) {
        if (j + 32 * KV_WAVES < S_)   // prefetch own next K tile
            __builtin_prefetch(krow0 + (size_t)(j + 32 * KV_WAVES) * D_, 0, 1);

        // ---- scores S = Q*K^T  (16 x 32, two 16x16 C tiles) ----
        v8f s0 = {}, s1 = {};
#pragma unroll
        for (int c = 0; c < 4; ++c) {
            v16bf bk0 = *(const v16bf*)(krow0 + (size_t)j * D_ + c * 32);
            v16bf bk1 = *(const v16bf*)(krow1 + (size_t)j * D_ + c * 32);
            s0 = __builtin_amdgcn_wmma_f32_16x16x32_bf16(false, qa[c], false, bk0, (short)0, s0, false, false);
            s1 = __builtin_amdgcn_wmma_f32_16x16x32_bf16(false, qa[c], false, bk1, (short)0, s1, false, false);
        }

        // ---- online softmax (rows M = r + 8*hi live across 16 lanes) ----
#pragma unroll
        for (int r = 0; r < 8; ++r) {
            float rv = fmaxf(s0[r], s1[r]);
#pragma unroll
            for (int mk = 1; mk < 16; mk <<= 1)
                rv = fmaxf(rv, __shfl_xor(rv, mk, 32));
            float mnew  = fmaxf(mrow[r], rv);
            float alpha = __expf(mrow[r] - mnew);
            float p0 = __expf(s0[r] - mnew);
            float p1 = __expf(s1[r] - mnew);
            float rs = p0 + p1;
#pragma unroll
            for (int mk = 1; mk < 16; mk <<= 1)
                rs += __shfl_xor(rs, mk, 32);
            lrow[r] = lrow[r] * alpha + rs;
            mrow[r] = mnew;
#pragma unroll
            for (int n = 0; n < 8; ++n) o[n][r] *= alpha;
            int Mr = r + 8 * hi;
            plds[wave][Mr * 32 + m]      = (__bf16)p0;
            plds[wave][Mr * 32 + 16 + m] = (__bf16)p1;
        }
        __syncthreads();

        // ---- P A-operand (16x32) from LDS via ds_load_b128 pairs ----
        v8bf plo = *(const v8bf*)&plds[wave][m * 32 + kb16];
        v8bf phi = *(const v8bf*)&plds[wave][m * 32 + kb16 + 8];
        v16bf pa = cat8(plo, phi);

#pragma unroll
        for (int n = 0; n < 8; ++n) {
            // B = V chunk 32x16: col = d = n*16+m ; K = kv row j+kb16+e
            v16bf bv = *(const v16bf*)(Vt + ((size_t)b * D_ + n * 16 + m) * S_ + j + kb16);
            o[n] = __builtin_amdgcn_wmma_f32_16x16x32_bf16(false, pa, false, bv, (short)0, o[n], false, false);
        }
        __syncthreads();
    }

    // ---- stage partial results to LDS ----
#pragma unroll
    for (int r = 0; r < 8; ++r)
#pragma unroll
        for (int n = 0; n < 8; ++n)
            ols[wave][r + 8 * hi][n * 16 + m] = o[n][r];
    if (m == 0) {
#pragma unroll
        for (int r = 0; r < 8; ++r) {
            mls[wave][0][r + 8 * hi] = mrow[r];
            mls[wave][1][r + 8 * hi] = lrow[r];
        }
    }
    __syncthreads();

    // ---- merge across waves: 128 threads -> (row, 16-col chunk) ----
    const int row = threadIdx.x >> 3;                    // 0..15
    const int c0  = (threadIdx.x & 7) * 16;              // 16 cols each
    float mg = -1e30f;
#pragma unroll
    for (int w = 0; w < KV_WAVES; ++w)
        mg = fmaxf(mg, mls[w][0][row]);
    float lg = 0.0f;
    float acc[16];
#pragma unroll
    for (int cc = 0; cc < 16; ++cc) acc[cc] = 0.0f;
#pragma unroll
    for (int w = 0; w < KV_WAVES; ++w) {
        float sc = __expf(mls[w][0][row] - mg);
        lg += mls[w][1][row] * sc;
#pragma unroll
        for (int cc = 0; cc < 16; ++cc)
            acc[cc] += sc * ols[w][row][c0 + cc];
    }
    float inv = 1.0f / lg;
    float* op = out + ((size_t)b * S_ + qt * 16 + row) * D_ + c0;
#pragma unroll
    for (int cc = 0; cc < 16; ++cc)
        op[cc] = acc[cc] * inv;
}

// ---------------------------------------------------------------------------
extern "C" void kernel_launch(void* const* d_in, const int* in_sizes, int n_in,
                              void* d_out, int out_size, void* d_ws, size_t ws_size,
                              hipStream_t stream)
{
    const float* x  = (const float*)d_in[0];
    const float* Wq = (const float*)d_in[1];
    const float* bq = (const float*)d_in[2];
    const float* Wk = (const float*)d_in[3];
    const float* bk = (const float*)d_in[4];
    const float* Wv = (const float*)d_in[5];
    const float* bv = (const float*)d_in[6];
    float* out = (float*)d_out;

    // workspace: Q | K | V^T (2MB each) then Wt_q | Wt_k | Wt_v (256KB each)
    const size_t qkv_elems = (size_t)B_ * S_ * D_;
    const size_t w_elems   = (size_t)E_ * D_;
    __bf16* Qw  = (__bf16*)d_ws;
    __bf16* Kw  = Qw + qkv_elems;
    __bf16* Vt  = Kw + qkv_elems;
    __bf16* Wtq = Vt + qkv_elems;
    __bf16* Wtk = Wtq + w_elems;
    __bf16* Wtv = Wtk + w_elems;

    const int wgrid = (int)((w_elems + 255) / 256);       // 512
    convert_w_kernel<<<wgrid, 256, 0, stream>>>(Wq, Wtq);
    convert_w_kernel<<<wgrid, 256, 0, stream>>>(Wk, Wtk);
    convert_w_kernel<<<wgrid, 256, 0, stream>>>(Wv, Wtv);

    const int row_tiles = (B_ * S_) / 16;                 // 512
    qkv_proj_kernel<<<row_tiles, 256, 0, stream>>>(x, Wtq, bq, Wtk, bk, Wtv, bv,
                                                   Qw, Kw, Vt);

    flash_attn_kernel<<<row_tiles, 32 * KV_WAVES, 0, stream>>>(Qw, Kw, Vt, out);
}
